// VisualHead_83468394431088
// MI455X (gfx1250) — matrix operand order repaired and checked by reading, not compiled, not run on hardware
//
#include <hip/hip_runtime.h>
#include <hip/hip_bf16.h>

// ---------------------------------------------------------------------------
// Problem constants (reference: B,C,V,D,K = 4096,1024,2000,300,5)
// ---------------------------------------------------------------------------
#define BB 4096
#define CC 1024
#define VV 2000
#define DD 300
#define KK 5

typedef __attribute__((ext_vector_type(16))) __bf16 v16bf;
typedef __attribute__((ext_vector_type(8)))  __bf16 bf16x8;
typedef __attribute__((ext_vector_type(8)))  float  v8f;
typedef __attribute__((ext_vector_type(4)))  float  f32x4;
typedef __attribute__((ext_vector_type(4)))  unsigned int   u32x4;
typedef __attribute__((ext_vector_type(4)))  unsigned short u16x4;

// CDNA5 async global->LDS path (ASYNCcnt) when the toolchain declares it.
#if defined(__has_builtin)
#if __has_builtin(__builtin_amdgcn_global_load_async_to_lds_b128) && \
    __has_builtin(__builtin_amdgcn_s_wait_asynccnt)
#define USE_ASYNC_LDS 1
#endif
#endif
#ifndef USE_ASYNC_LDS
#define USE_ASYNC_LDS 0
#endif

#if USE_ASYNC_LDS
// Builtin signature (from hipcc diagnostic): params are v4i pointers in
// addrspace(1) (global src) and addrspace(3) (LDS dst), plus imm offset/cpol.
typedef int v4i_t __attribute__((__vector_size__(16)));
typedef __attribute__((address_space(1))) v4i_t* as1_v4i;
typedef __attribute__((address_space(3))) v4i_t* as3_v4i;
#endif

__device__ __forceinline__ unsigned short f2bf(float f) {
    unsigned u = __float_as_uint(f);
    u += 0x7FFFu + ((u >> 16) & 1u);      // round-to-nearest-even
    return (unsigned short)(u >> 16);
}

// ---------------------------------------------------------------------------
// fp32 -> bf16 bulk convert (vectorized, 4 elems/thread)
// ---------------------------------------------------------------------------
__global__ void cvt_bf16_kernel(const float* __restrict__ src,
                                unsigned short* __restrict__ dst, int n4) {
    int i = blockIdx.x * blockDim.x + threadIdx.x;
    if (i >= n4) return;
    f32x4 v = *(const f32x4*)(src + (size_t)i * 4);
    u16x4 o;
    o.x = f2bf(v.x); o.y = f2bf(v.y); o.z = f2bf(v.z); o.w = f2bf(v.w);
    *(u16x4*)(dst + (size_t)i * 4) = o;
}

// ---------------------------------------------------------------------------
// e[v,c] = dot(tab[v,:D], Wm[c,:D]) + bm[c]   (fp32, small: 1.2 GF)
// grid = (C/256, V), block = 256. Tab row staged in LDS.
// ---------------------------------------------------------------------------
__global__ __launch_bounds__(256)
void emb_map_kernel(const float* __restrict__ tab, const float* __restrict__ Wm,
                    const float* __restrict__ bm, float* __restrict__ e) {
    __shared__ float srow[304];
    const int v = blockIdx.y;
    const int c = blockIdx.x * 256 + threadIdx.x;
    for (int d = threadIdx.x; d < DD; d += 256) srow[d] = tab[(size_t)v * DD + d];
    __syncthreads();
    const float* wr = Wm + (size_t)c * DD;
    float s = 0.f;
    int d = 0;
    #pragma unroll 4
    for (; d + 4 <= DD; d += 4)
        s += srow[d] * wr[d] + srow[d + 1] * wr[d + 1] +
             srow[d + 2] * wr[d + 2] + srow[d + 3] * wr[d + 3];
    e[(size_t)v * CC + c] = s + bm[c];
}

// ---------------------------------------------------------------------------
// Top-4 (excluding label) per row via 4 wave-wide argmax passes.
// One wave32 per row; deterministic smallest-index tie break.
// ---------------------------------------------------------------------------
__global__ void topk_kernel(const float* __restrict__ out1,
                            const int* __restrict__ label,
                            int* __restrict__ topk, float* __restrict__ topk_f) {
    const int b    = blockIdx.x;
    const int lane = threadIdx.x;
    const float* row = out1 + (size_t)b * VV;
    const int lab = label[b];
    int sel[4] = {-1, -1, -1, -1};
    for (int f = 0; f < 4; ++f) {
        float bv = -3.4e38f;
        int   bi = 0x7fffffff;
        for (int i = lane; i < VV; i += 32) {
            if (i == lab) continue;
            bool ex = false;
            #pragma unroll
            for (int j = 0; j < 4; ++j) ex |= (j < f) && (sel[j] == i);
            if (ex) continue;
            float val = row[i];
            if (val > bv || (val == bv && i < bi)) { bv = val; bi = i; }
        }
        #pragma unroll
        for (int off = 16; off > 0; off >>= 1) {
            float ov = __shfl_xor(bv, off, 32);
            int   oi = __shfl_xor(bi, off, 32);
            if (ov > bv || (ov == bv && oi < bi)) { bv = ov; bi = oi; }
        }
        sel[f] = bi;   // uniform across lanes after reduction
    }
    if (lane == 0) {
        topk[b * KK] = lab;
        topk_f[b * KK] = (float)lab;
        #pragma unroll
        for (int f = 0; f < 4; ++f) {
            topk[b * KK + 1 + f]   = sel[f];
            topk_f[b * KK + 1 + f] = (float)sel[f];
        }
    }
}

// ---------------------------------------------------------------------------
// Ffused_bf16[bk, c] = bf16( x[b,c] + e[topk[bk], c] )   (4 elems/thread)
// ---------------------------------------------------------------------------
__global__ void fuse_kernel(const float* __restrict__ x, const float* __restrict__ e,
                            const int* __restrict__ topk,
                            unsigned short* __restrict__ F) {
    const int c4    = CC >> 2;
    const int t     = blockIdx.x * blockDim.x + threadIdx.x;
    const int total = BB * KK * c4;
    if (t >= total) return;
    const int c  = (t % c4) * 4;
    const int bk = t / c4;
    const int b  = bk / KK;
    const int v  = topk[bk];
    f32x4 xv = *(const f32x4*)(x + (size_t)b * CC + c);
    f32x4 ev = *(const f32x4*)(e + (size_t)v * CC + c);
    u16x4 o;
    o.x = f2bf(xv.x + ev.x); o.y = f2bf(xv.y + ev.y);
    o.z = f2bf(xv.z + ev.z); o.w = f2bf(xv.w + ev.w);
    *(u16x4*)(F + (size_t)bk * CC + c) = o;
}

// ---------------------------------------------------------------------------
// bf16 GEMM, NT form:  C[M,N] = A[M,K] * Bw[N,K]^T + bias[N], fp32 accum.
// Block 128 threads (4 waves), tile 128x128, wave tile 64x64 (4x4 WMMA accums).
// K staged 64 at a time, double-buffered LDS, one barrier per stage.
// Global->LDS via async DMA (ASYNCcnt) when available.
// LDS row stride 72 bf16 (36 dw, gcd 4 with 64 banks) -> conflict-free b128.
// Requires: M % 128 == 0, K % 64 == 0 (N edge handled).
// ---------------------------------------------------------------------------
#define TM 128
#define TN 128
#define TK 64
#define LDT 72

__global__ __launch_bounds__(128)
void gemm_bf16_nt(const unsigned short* __restrict__ A,
                  const unsigned short* __restrict__ Bw,
                  const float* __restrict__ bias,
                  float* __restrict__ C, int M, int N, int K) {
    __shared__ unsigned short sA[2][TM * LDT];
    __shared__ unsigned short sB[2][TN * LDT];

    const int tid   = threadIdx.x;
    const int lane  = tid & 31;
    const int wave  = tid >> 5;           // 0..3
    const int mg    = wave >> 1;          // 0..1 : 64-row slice
    const int ng    = wave & 1;           // 0..1 : 64-col slice
    const int l16   = lane & 15;
    const int lhalf = lane >> 4;          // 0/1
    const int rowBase = blockIdx.y * TM;
    const int colBase = blockIdx.x * TN;

    // loader: 16B (8 bf16)/thread/pass; 128x64 tile = 8 passes
    const int lr = tid >> 3;              // 0..15
    const int lc = (tid & 7) * 8;         // 0,8,..,56

    // Pre-zero out-of-range B rows (N edge) in both buffers; those slots are
    // never written again, so they stay zero for every K stage.
    #pragma unroll
    for (int p = 0; p < 8; ++p) {
        const int r = p * 16 + lr;
        if (colBase + r >= N) {
            u32x4 z = {};
            *(u32x4*)(&sB[0][r * LDT + lc]) = z;
            *(u32x4*)(&sB[1][r * LDT + lc]) = z;
        }
    }

    auto load_tile = [&](int buf, int k0) {
        #pragma unroll
        for (int p = 0; p < 8; ++p) {
            const int r  = p * 16 + lr;
            const int bn = colBase + r;
            const unsigned short* ga = A  + (size_t)(rowBase + r) * K + k0 + lc;
            const unsigned short* gb = Bw + (size_t)bn * K + k0 + lc;
            unsigned short* la = &sA[buf][r * LDT + lc];
            unsigned short* lb = &sB[buf][r * LDT + lc];
#if USE_ASYNC_LDS
            __builtin_amdgcn_global_load_async_to_lds_b128(
                (as1_v4i)ga, (as3_v4i)la, 0, 0);
            if (bn < N)
                __builtin_amdgcn_global_load_async_to_lds_b128(
                    (as1_v4i)gb, (as3_v4i)lb, 0, 0);
#else
            u32x4 va = *(const u32x4*)ga;
            *(u32x4*)la = va;
            if (bn < N) {
                u32x4 vb = *(const u32x4*)gb;
                *(u32x4*)lb = vb;
            }
#endif
        }
    };

    v8f acc[4][4] = {};

    load_tile(0, 0);                      // prologue: stage 0 -> buf 0
    const int S = K / TK;
    for (int s = 0; s < S; ++s) {
        const int cur = s & 1;
#if USE_ASYNC_LDS
        __builtin_amdgcn_s_wait_asynccnt(0);
#endif
        __syncthreads();                  // publish stage s, retire stage s-1 reads
        if (s + 1 < S) load_tile(cur ^ 1, (s + 1) * TK);

        const unsigned short* bufA = sA[cur];
        const unsigned short* bufB = sB[cur];
        #pragma unroll
        for (int kk = 0; kk < TK; kk += 32) {
            // A fragment 16x32: lane row l16, K = {kh..kh+7, kh+16..kh+23}
            v16bf aF[4];
            #pragma unroll
            for (int m = 0; m < 4; ++m) {
                const int r = mg * 64 + m * 16 + l16;
                bf16x8 lo = *(const bf16x8*)(&bufA[r * LDT + kk + lhalf * 8]);
                bf16x8 hi = *(const bf16x8*)(&bufA[r * LDT + kk + lhalf * 8 + 16]);
                aF[m] = __builtin_shufflevector(lo, hi,
                        0,1,2,3,4,5,6,7,8,9,10,11,12,13,14,15);
            }
            // B fragment 32x16: lane column l16, K = {kh..kh+7, kh+16..kh+23}
            v16bf bF[4];
            #pragma unroll
            for (int n = 0; n < 4; ++n) {
                const int cN = ng * 64 + n * 16 + l16;
                bf16x8 lo = *(const bf16x8*)(&bufB[cN * LDT + kk + lhalf * 8]);
                bf16x8 hi = *(const bf16x8*)(&bufB[cN * LDT + kk + lhalf * 8 + 16]);
                bF[n] = __builtin_shufflevector(lo, hi,
                        0,1,2,3,4,5,6,7,8,9,10,11,12,13,14,15);
            }
            #pragma unroll
            for (int m = 0; m < 4; ++m)
                #pragma unroll
                for (int n = 0; n < 4; ++n)
                    acc[m][n] = __builtin_amdgcn_wmma_f32_16x16x32_bf16(
                        false, aF[m], false, bF[n], (short)0, acc[m][n],
                        false, false);
        }
    }

    // epilogue: VGPR v -> M = v + 8*lhalf, N = l16
    #pragma unroll
    for (int n = 0; n < 4; ++n) {
        const int col = colBase + ng * 64 + n * 16 + l16;
        if (col >= N) continue;
        const float bv = bias[col];
        #pragma unroll
        for (int m = 0; m < 4; ++m) {
            const int r0 = rowBase + mg * 64 + m * 16 + lhalf * 8;
            #pragma unroll
            for (int v = 0; v < 8; ++v)
                C[(size_t)(r0 + v) * N + col] = acc[m][n][v] + bv;
        }
    }
}

// ---------------------------------------------------------------------------
// Launcher
// ---------------------------------------------------------------------------
extern "C" void kernel_launch(void* const* d_in, const int* in_sizes, int n_in,
                              void* d_out, int out_size, void* d_ws, size_t ws_size,
                              hipStream_t stream) {
    const float* x     = (const float*)d_in[0];   // [B,C]
    const int*   label = (const int*)  d_in[1];   // [B]
    const float* W1    = (const float*)d_in[2];   // [V,C]
    const float* b1    = (const float*)d_in[3];   // [V]
    const float* tab   = (const float*)d_in[4];   // [V,D]
    const float* Wm    = (const float*)d_in[5];   // [C,D]
    const float* bm    = (const float*)d_in[6];   // [C]
    const float* W2    = (const float*)d_in[7];   // [V,C]
    const float* b2    = (const float*)d_in[8];   // [V]

    float* out1   = (float*)d_out;                                  // B*V
    float* out2   = out1 + (size_t)BB * VV;                         // B*K*V
    float* outIdx = out2 + (size_t)BB * KK * VV;                    // B*K (as float)

    // workspace carve-out (256B aligned)
    size_t off = 0;
    auto carve = [&](size_t bytes) {
        void* p = (char*)d_ws + off;
        off = (off + bytes + 255) & ~(size_t)255;
        return p;
    };
    unsigned short* xb   = (unsigned short*)carve((size_t)BB * CC * 2);      // 8 MB
    unsigned short* W1b  = (unsigned short*)carve((size_t)VV * CC * 2);      // 4 MB
    unsigned short* W2b  = (unsigned short*)carve((size_t)VV * CC * 2);      // 4 MB
    float*          e    = (float*)         carve((size_t)VV * CC * 4);      // 8 MB
    unsigned short* Ff   = (unsigned short*)carve((size_t)BB * KK * CC * 2); // 40 MB
    int*            topk = (int*)           carve((size_t)BB * KK * 4);      // 80 KB

    // 1) fp32 -> bf16 operands
    {
        int n4 = (BB * CC) / 4;
        cvt_bf16_kernel<<<(n4 + 255) / 256, 256, 0, stream>>>(x, xb, n4);
        n4 = (VV * CC) / 4;
        cvt_bf16_kernel<<<(n4 + 255) / 256, 256, 0, stream>>>(W1, W1b, n4);
        cvt_bf16_kernel<<<(n4 + 255) / 256, 256, 0, stream>>>(W2, W2b, n4);
    }

    // 2) e = tab @ Wm^T + bm  (fp32, small)
    emb_map_kernel<<<dim3(CC / 256, VV), 256, 0, stream>>>(tab, Wm, bm, e);

    // 3) out1 = x @ W1^T + b1  (bf16 WMMA, fp32 accum)
    gemm_bf16_nt<<<dim3((VV + TN - 1) / TN, BB / TM), 128, 0, stream>>>(
        xb, W1b, b1, out1, BB, VV, CC);

    // 4) top-4 (label excluded) per row
    topk_kernel<<<BB, 32, 0, stream>>>(out1, label, topk, outIdx);

    // 5) Ffused = bf16(x + e[topk])
    {
        int total = BB * KK * (CC / 4);
        fuse_kernel<<<(total + 255) / 256, 256, 0, stream>>>(x, e, topk, Ff);
    }

    // 6) out2 = Ffused @ W2^T + b2  (dominant GEMM, 84 GF bf16 WMMA)
    gemm_bf16_nt<<<dim3((VV + TN - 1) / TN, (BB * KK) / TM), 128, 0, stream>>>(
        Ff, W2b, b2, out2, BB * KK, VV, CC);
}
